// LSTM2_29326036697974
// MI455X (gfx1250) — matrix operand order repaired and checked by reading, not compiled, not run on hardware
//
#include <hip/hip_runtime.h>
#include <hip/hip_bf16.h>
#include <stdint.h>

// ---------------------------------------------------------------------------
// Problem constants (from reference)
// ---------------------------------------------------------------------------
#define BB   8192
#define II   128
#define TT   21
#define H1C  256
#define H2C  256
#define H3C  128
#define MM   (BB * TT)          // 172032 rows, divisible by 64

typedef __attribute__((ext_vector_type(16))) __bf16 v16bf;
typedef __attribute__((ext_vector_type(8)))  __bf16 v8bf;
typedef __attribute__((ext_vector_type(8)))  float  v8f;

// ---------------------------------------------------------------------------
// Helpers
// ---------------------------------------------------------------------------
__device__ __forceinline__ float fsigmoid(float x) {
    return 1.0f / (1.0f + __expf(-x));
}
__device__ __forceinline__ float ftanh(float x) {
    // tanh(x) = 2*sigmoid(2x) - 1  (uses v_exp transcendental)
    return 2.0f / (1.0f + __expf(-2.0f * x)) - 1.0f;
}

// ---------------------------------------------------------------------------
// 1) Transpose + convert: x (B, 128, 21) f32 -> A1 (B*21, 128) bf16
//    One block per batch element, staged through LDS for coalescing.
// ---------------------------------------------------------------------------
__global__ __launch_bounds__(256)
void transpose_cvt_kernel(const float* __restrict__ x, __bf16* __restrict__ out) {
    __shared__ float tile[II * TT];           // 128*21 = 2688 floats (10.5 KB)
    const int b = blockIdx.x;
    const float* xb = x + (size_t)b * (II * TT);
    for (int idx = threadIdx.x; idx < II * TT; idx += 256)
        tile[idx] = xb[idx];                  // coalesced read
    __syncthreads();
    __bf16* ob = out + (size_t)b * TT * II;
    for (int idx = threadIdx.x; idx < II * TT; idx += 256) {
        const int t = idx >> 7;               // idx / 128
        const int i = idx & 127;              // idx % 128
        ob[(size_t)t * II + i] = (__bf16)tile[i * TT + t];  // coalesced write
    }
}

// ---------------------------------------------------------------------------
// 2) Generic f32 -> bf16 convert (weights)
// ---------------------------------------------------------------------------
__global__ void cvt_bf16_kernel(const float* __restrict__ src,
                                __bf16* __restrict__ dst, int n) {
    int i = blockIdx.x * blockDim.x + threadIdx.x;
    if (i < n) dst[i] = (__bf16)src[i];
}

// 3) bias = b_ih + b_hh (kept in f32 for the epilogue)
__global__ void bias_add_kernel(const float* __restrict__ a,
                                const float* __restrict__ b,
                                float* __restrict__ out, int n) {
    int i = blockIdx.x * blockDim.x + threadIdx.x;
    if (i < n) out[i] = a[i] + b[i];
}

// ---------------------------------------------------------------------------
// 4) Fused LSTM layer: Hout = sigmoid(o) * tanh(sigmoid(i) * tanh(g))
//    gates = A @ W^T + bias ; f-gate skipped (f * c0 == 0).
//    A: [M, KTOT] bf16 row-major.  W: [4*HOUT, KTOT] bf16 row-major.
//    One wave -> MT vertically-stacked 16x16 output tiles sharing the same
//    B fragments (3 gates), i.e. 3*MT WMMA accumulators. Per k-step:
//    2*MT A-loads + 6 B-loads feed 3*MT independent WMMAs.
//    Grid: (M/(16*MT), HOUT/128), block 256 = 8 waves, wave w -> N-tile w.
// ---------------------------------------------------------------------------
template <int KTOT, int HOUT, int MT>
__global__ __launch_bounds__(256)
void lstm_layer_kernel(const __bf16* __restrict__ A,
                       const __bf16* __restrict__ W,
                       const float*  __restrict__ bias,
                       __bf16* __restrict__ Hout) {
    const int lane   = threadIdx.x & 31;
    const int wave   = threadIdx.x >> 5;
    const int lane16 = lane & 15;
    const int laneHi = lane >> 4;

    const int mbase = blockIdx.x * (16 * MT);                // tile-block row base
    const int nbase = (blockIdx.y * 8 + wave) * 16;          // tile col base (in H)
    const int ncol  = nbase + lane16;                        // this lane's column

    // A-fragment source rows for this lane (rows m = mbase + j*16 + lane16)
    const __bf16* arow[MT];
#pragma unroll
    for (int j = 0; j < MT; ++j)
        arow[j] = A + (size_t)(mbase + j * 16 + lane16) * KTOT;

    // B-fragment source rows: W rows are gate columns. Gates i, g, o.
    const __bf16* wi = W + (size_t)(0 * HOUT + ncol) * KTOT;
    const __bf16* wg = W + (size_t)(2 * HOUT + ncol) * KTOT;
    const __bf16* wo = W + (size_t)(3 * HOUT + ncol) * KTOT;

    v8f ci[MT], cg[MT], co[MT];
#pragma unroll
    for (int j = 0; j < MT; ++j) { ci[j] = (v8f){}; cg[j] = (v8f){}; co[j] = (v8f){}; }

#pragma unroll
    for (int k = 0; k < KTOT; k += 32) {
        // ---- B fragments (32x16 bf16): lane n holds B[k+hi*16 .. +15][n],
        //      which is 32 contiguous bytes of W row `ncol`.
        v16bf bi = *(const v16bf*)(wi + k + laneHi * 16);
        v16bf bg = *(const v16bf*)(wg + k + laneHi * 16);
        v16bf bo = *(const v16bf*)(wo + k + laneHi * 16);

        // ---- A fragments (16x32 bf16): lane holds K = k+hi*16+[0..15]
        //      (contiguous, matching the B fragments)
        v16bf a[MT];
#pragma unroll
        for (int j = 0; j < MT; ++j) {
            a[j] = *(const v16bf*)(arow[j] + k + laneHi * 16);
        }

        // ---- 3*MT independent WMMAs (no D->A/B hazards between them)
#pragma unroll
        for (int j = 0; j < MT; ++j) {
            ci[j] = __builtin_amdgcn_wmma_f32_16x16x32_bf16(false, a[j], false, bi,
                                                            (short)0, ci[j], false, false);
            cg[j] = __builtin_amdgcn_wmma_f32_16x16x32_bf16(false, a[j], false, bg,
                                                            (short)0, cg[j], false, false);
            co[j] = __builtin_amdgcn_wmma_f32_16x16x32_bf16(false, a[j], false, bo,
                                                            (short)0, co[j], false, false);
        }
    }

    // Per-column bias (combined b_ih + b_hh), gate order i,f,g,o
    const float bii = bias[0 * HOUT + ncol];
    const float big = bias[2 * HOUT + ncol];
    const float bio = bias[3 * HOUT + ncol];

    // C/D layout: element r -> row (tile base) + laneHi*8 + r, col = ncol
#pragma unroll
    for (int j = 0; j < MT; ++j) {
#pragma unroll
        for (int r = 0; r < 8; ++r) {
            const float iv = fsigmoid(ci[j][r] + bii);
            const float gv = ftanh(cg[j][r] + big);
            const float ov = fsigmoid(co[j][r] + bio);
            const float h  = ov * ftanh(iv * gv);
            Hout[(size_t)(mbase + j * 16 + laneHi * 8 + r) * HOUT + ncol] = (__bf16)h;
        }
    }
}

// ---------------------------------------------------------------------------
// 5) fc1: y[m] = dot(h3[m, 0:128], fc1_w) + fc1_b   (memory-trivial GEMV)
// ---------------------------------------------------------------------------
__global__ __launch_bounds__(256)
void fc1_kernel(const __bf16* __restrict__ h3, const float* __restrict__ w,
                const float* __restrict__ b, float* __restrict__ y) {
    const int m = blockIdx.x * blockDim.x + threadIdx.x;
    if (m >= MM) return;
    const __bf16* row = h3 + (size_t)m * H3C;
    float acc = 0.0f;
#pragma unroll
    for (int j = 0; j < H3C; ++j) acc += (float)row[j] * w[j];
    y[m] = acc + b[0];
}

// ---------------------------------------------------------------------------
// 6) fc2: out[b, t'] = sum_t y[b,t] * fc2_w[t', t] + fc2_b[t']
// ---------------------------------------------------------------------------
__global__ __launch_bounds__(256)
void fc2_kernel(const float* __restrict__ y, const float* __restrict__ w,
                const float* __restrict__ b, float* __restrict__ out) {
    const int idx = blockIdx.x * blockDim.x + threadIdx.x;
    if (idx >= BB * TT) return;
    const int bi = idx / TT;
    const int tp = idx - bi * TT;
    const float* yr = y + (size_t)bi * TT;
    float acc = b[tp];
#pragma unroll
    for (int t = 0; t < TT; ++t) acc += yr[t] * w[tp * TT + t];
    out[idx] = acc;
}

// ---------------------------------------------------------------------------
// Host side
// ---------------------------------------------------------------------------
static inline size_t align256(size_t v) { return (v + 255) & ~(size_t)255; }

extern "C" void kernel_launch(void* const* d_in, const int* in_sizes, int n_in,
                              void* d_out, int out_size, void* d_ws, size_t ws_size,
                              hipStream_t stream) {
    const float* x     = (const float*)d_in[0];
    const float* W1    = (const float*)d_in[1];
    const float* b1ih  = (const float*)d_in[2];
    const float* b1hh  = (const float*)d_in[3];
    const float* W2    = (const float*)d_in[4];
    const float* b2ih  = (const float*)d_in[5];
    const float* b2hh  = (const float*)d_in[6];
    const float* W3    = (const float*)d_in[7];
    const float* b3ih  = (const float*)d_in[8];
    const float* b3hh  = (const float*)d_in[9];
    const float* fc1w  = (const float*)d_in[10];
    const float* fc1b  = (const float*)d_in[11];
    const float* fc2w  = (const float*)d_in[12];
    const float* fc2b  = (const float*)d_in[13];
    float* out = (float*)d_out;

    // Workspace carve-up (ping-pong activation buffers, bf16)
    char* ws = (char*)d_ws;
    size_t off = 0;
    const size_t bufBytes = (size_t)MM * 256 * sizeof(__bf16);   // 88 MB
    __bf16* buf0 = (__bf16*)(ws + off); off = align256(off + bufBytes);
    __bf16* buf1 = (__bf16*)(ws + off); off = align256(off + bufBytes);
    __bf16* W1b  = (__bf16*)(ws + off); off = align256(off + (size_t)4 * H1C * II  * 2);
    __bf16* W2b  = (__bf16*)(ws + off); off = align256(off + (size_t)4 * H2C * H1C * 2);
    __bf16* W3b  = (__bf16*)(ws + off); off = align256(off + (size_t)4 * H3C * H2C * 2);
    float* bias1 = (float*)(ws + off);  off = align256(off + (size_t)4 * H1C * 4);
    float* bias2 = (float*)(ws + off);  off = align256(off + (size_t)4 * H2C * 4);
    float* bias3 = (float*)(ws + off);  off = align256(off + (size_t)4 * H3C * 4);
    float* yvec  = (float*)(ws + off);  off = align256(off + (size_t)MM * 4);
    (void)ws_size; (void)n_in; (void)in_sizes; (void)out_size;

    // ---- Stage 0: convert inputs
    transpose_cvt_kernel<<<BB, 256, 0, stream>>>(x, buf0);            // A1 in buf0
    cvt_bf16_kernel<<<(4 * H1C * II  + 255) / 256, 256, 0, stream>>>(W1, W1b, 4 * H1C * II);
    cvt_bf16_kernel<<<(4 * H2C * H1C + 255) / 256, 256, 0, stream>>>(W2, W2b, 4 * H2C * H1C);
    cvt_bf16_kernel<<<(4 * H3C * H2C + 255) / 256, 256, 0, stream>>>(W3, W3b, 4 * H3C * H2C);
    bias_add_kernel<<<(4 * H1C + 255) / 256, 256, 0, stream>>>(b1ih, b1hh, bias1, 4 * H1C);
    bias_add_kernel<<<(4 * H2C + 255) / 256, 256, 0, stream>>>(b2ih, b2hh, bias2, 4 * H2C);
    bias_add_kernel<<<(4 * H3C + 255) / 256, 256, 0, stream>>>(b3ih, b3hh, bias3, 4 * H3C);

    // ---- Stage 1..3: fused LSTM layers (WMMA bf16, f-gate skipped)
    constexpr int MT = 4;                      // M-tiles per wave (register blocking)
    {
        dim3 grid(MM / (16 * MT), H1C / 128);
        lstm_layer_kernel<II, H1C, MT><<<grid, 256, 0, stream>>>(buf0, W1b, bias1, buf1);
    }
    {
        dim3 grid(MM / (16 * MT), H2C / 128);
        lstm_layer_kernel<H1C, H2C, MT><<<grid, 256, 0, stream>>>(buf1, W2b, bias2, buf0);
    }
    {
        dim3 grid(MM / (16 * MT), H3C / 128);
        lstm_layer_kernel<H2C, H3C, MT><<<grid, 256, 0, stream>>>(buf0, W3b, bias3, buf1);
    }

    // ---- Stage 4: fc1 GEMV + fc2 (21x21)
    fc1_kernel<<<(MM + 255) / 256, 256, 0, stream>>>(buf1, fc1w, fc1b, yvec);
    fc2_kernel<<<(BB * TT + 255) / 256, 256, 0, stream>>>(yvec, fc2w, fc2b, out);
}
